// GeneratorModel_23467701305950
// MI455X (gfx1250) — matrix operand order repaired and checked by reading, not compiled, not run on hardware
//
#include <hip/hip_runtime.h>
#include <stdint.h>
#include <stddef.h>

// ---------------------------------------------------------------------------
// Types / helpers
// ---------------------------------------------------------------------------
typedef __bf16 bf16;
typedef __attribute__((ext_vector_type(16))) __bf16 bf16x16;
typedef __attribute__((ext_vector_type(8)))  float  v8f;

union Frag { unsigned int u[8]; bf16x16 v; };

__device__ __forceinline__ float bf2f(bf16 x) {
  unsigned short s = __builtin_bit_cast(unsigned short, x);
  unsigned int u = ((unsigned int)s) << 16;
  return __builtin_bit_cast(float, u);
}
__device__ __forceinline__ bf16 f2bf(float f) {
  unsigned int u = __builtin_bit_cast(unsigned int, f);
  u += 0x7FFFu + ((u >> 16) & 1u);           // round-to-nearest-even
  unsigned short s = (unsigned short)(u >> 16);
  return __builtin_bit_cast(bf16, s);
}

static const int Bn = 2048;   // batch

// ---------------------------------------------------------------------------
// f32 -> bf16 weight conversion
// ---------------------------------------------------------------------------
__global__ void k_f2bf(const float* __restrict__ in, bf16* __restrict__ out, int n) {
  int i = blockIdx.x * blockDim.x + threadIdx.x;
  if (i < n) out[i] = f2bf(in[i]);
}

// ---------------------------------------------------------------------------
// conv1x1 (Cin -> 4) + ReLU, output bf16.  x[b,o,l] layout (B,4,L)
// ---------------------------------------------------------------------------
__global__ void k_conv1x1_relu(const float* __restrict__ s,
                               const float* __restrict__ w,
                               const float* __restrict__ b,
                               bf16* __restrict__ x, int Cin, int L, int total) {
  int idx = blockIdx.x * blockDim.x + threadIdx.x;
  if (idx >= total) return;
  int bb = idx / L, l = idx - bb * L;
  const float* srow = s + (size_t)bb * Cin * L + l;
  float acc[4];
  #pragma unroll
  for (int o = 0; o < 4; ++o) acc[o] = b[o];
  for (int i = 0; i < Cin; ++i) {
    float xv = srow[(size_t)i * L];
    #pragma unroll
    for (int o = 0; o < 4; ++o) acc[o] += xv * w[o * Cin + i];
  }
  bf16* orow = x + (size_t)bb * 4 * L + l;
  #pragma unroll
  for (int o = 0; o < 4; ++o) orow[(size_t)o * L] = f2bf(acc[o] > 0.f ? acc[o] : 0.f);
}

// ---------------------------------------------------------------------------
// conv5 stage 1: (Cin -> 2), pad 2, + ReLU.  w layout (2, Cin, 5)
// ---------------------------------------------------------------------------
__global__ void k_conv5_ac1(const bf16* __restrict__ x,
                            const float* __restrict__ w,
                            const float* __restrict__ b,
                            bf16* __restrict__ h1, int Cin, int L, int total) {
  int idx = blockIdx.x * blockDim.x + threadIdx.x;
  if (idx >= total) return;
  int bb = idx / L, l = idx - bb * L;
  const bf16* xb = x + (size_t)bb * Cin * L;
  float a0 = b[0], a1 = b[1];
  for (int i = 0; i < Cin; ++i) {
    #pragma unroll
    for (int t = 0; t < 5; ++t) {
      int ll = l - 2 + t;
      if (ll < 0 || ll >= L) continue;
      float xv = bf2f(xb[(size_t)i * L + ll]);
      a0 += xv * w[(0 * Cin + i) * 5 + t];
      a1 += xv * w[(1 * Cin + i) * 5 + t];
    }
  }
  bf16* o = h1 + (size_t)bb * 2 * L + l;
  o[0]         = f2bf(a0 > 0.f ? a0 : 0.f);
  o[(size_t)L] = f2bf(a1 > 0.f ? a1 : 0.f);
}

// ---------------------------------------------------------------------------
// conv5 stage 2: (2 -> Cout), pad 2, no activation.  w layout (Cout, 2, 5)
// ---------------------------------------------------------------------------
__global__ void k_conv5_ac2(const bf16* __restrict__ h1,
                            const float* __restrict__ w,
                            const float* __restrict__ b,
                            bf16* __restrict__ h2, int Cout, int L, int total) {
  int idx = blockIdx.x * blockDim.x + threadIdx.x;
  if (idx >= total) return;
  int bb = idx / L, l = idx - bb * L;
  const bf16* hb = h1 + (size_t)bb * 2 * L;
  float acc[8];
  for (int o = 0; o < Cout; ++o) acc[o] = b[o];
  #pragma unroll
  for (int i = 0; i < 2; ++i) {
    #pragma unroll
    for (int t = 0; t < 5; ++t) {
      int ll = l - 2 + t;
      if (ll < 0 || ll >= L) continue;
      float xv = bf2f(hb[(size_t)i * L + ll]);
      for (int o = 0; o < Cout; ++o) acc[o] += xv * w[(o * 2 + i) * 5 + t];
    }
  }
  bf16* orow = h2 + (size_t)bb * Cout * L + l;
  for (int o = 0; o < Cout; ++o) orow[(size_t)o * L] = f2bf(acc[o]);
}

// ---------------------------------------------------------------------------
// maxpool window 3 stride 2 pad 1 on (B, C, L) -> (B, C, L/2)
// ---------------------------------------------------------------------------
__global__ void k_maxpool(const bf16* __restrict__ x, bf16* __restrict__ y,
                          int C, int L, int total) {
  int idx = blockIdx.x * blockDim.x + threadIdx.x;
  if (idx >= total) return;
  int Lo = L >> 1;
  int lo = idx % Lo;
  int r  = idx / Lo;
  int c  = r % C;
  int bb = r / C;
  const bf16* row = x + ((size_t)bb * C + c) * L;
  float m = -__builtin_inff();
  #pragma unroll
  for (int t = 0; t < 3; ++t) {
    int l = 2 * lo - 1 + t;
    if (l < 0 || l >= L) continue;
    float v = bf2f(row[l]);
    m = v > m ? v : m;
  }
  y[((size_t)bb * C + c) * Lo + lo] = f2bf(m);
}

// ---------------------------------------------------------------------------
// Per-channel GEMM via V_WMMA_F32_16X16X32_BF16:
//   out[m, n] = sum_k A[m, k] * W[n, k] + bias[n]     (per channel c = blockIdx.z)
// Block = 256 threads = 8 waves.  Block tile 128(M) x 64(N), K-step 64.
// Per wave: 32Mx32N (4 accumulators) -> 8 WMMAs per K-step, 2 LDS b128/WMMA.
// Double-buffered LDS filled by GLOBAL_LOAD_ASYNC_TO_LDS_B128 (ASYNCcnt),
// pipelined one K-tile ahead; last iteration peeled (branch-free hot loop).
// ---------------------------------------------------------------------------
#define LDSROW 72                      // 64 K-elems padded to 72 (144B rows)
#define ATILE_E (128 * LDSROW)         // A tile buffer elems
#define BTILE_E (64 * LDSROW)          // B tile buffer elems

__global__ void __launch_bounds__(256)
k_bmm_wmma(const bf16* __restrict__ A, int aBatch, int aChan,
           const bf16* __restrict__ W, int wChan,
           const float* __restrict__ bias, int biasChan,
           bf16* __restrict__ out, int oBatch, int oChan,
           int K, int N) {
  __shared__ __align__(16) bf16 sA[2 * ATILE_E];   // 2 x 18KB
  __shared__ __align__(16) bf16 sB[2 * BTILE_E];   // 2 x 9KB

  const int c = blockIdx.z;
  const bf16*  Ab = A    + (size_t)c * aChan;
  const bf16*  Wb = W    + (size_t)c * wChan;
  const float* Bb = bias + (size_t)c * biasChan;
  bf16*        Ob = out  + (size_t)c * oChan;

  const int m0 = blockIdx.x * 128;
  const int n0 = blockIdx.y * 64;

  const int t    = threadIdx.x;
  const int lane = t & 31;
  const int wv   = t >> 5;      // wave 0..7
  const int wm   = wv & 3;      // M sub-tile 0..3 (32 rows each)
  const int wn   = wv >> 2;     // N half 0..1    (32 cols each)
  const int lr   = lane & 15;
  const int lh   = lane >> 4;

  // Staging: A tile = 128 rows x 8 chunks(16B) = 1024 chunks (4/thread),
  //          B tile =  64 rows x 8 chunks      =  512 chunks (2/thread).
  // => 6 async-to-LDS b128 per wave per K-tile.
  auto stage = [&](int k0, int buf) {
    bf16* la = sA + buf * ATILE_E;
    bf16* lb = sB + buf * BTILE_E;
    #pragma unroll
    for (int h = 0; h < 4; ++h) {
      const int ch = t + h * 256;
      const int r = ch >> 3, kc = ch & 7;
      unsigned long long ga = (unsigned long long)(Ab + (size_t)(m0 + r) * aBatch + k0 + kc * 8);
      unsigned lds = (unsigned)(size_t)(la + r * LDSROW + kc * 8);
      asm volatile("global_load_async_to_lds_b128 %0, %1, off" :: "v"(lds), "v"(ga) : "memory");
    }
    #pragma unroll
    for (int h = 0; h < 2; ++h) {
      const int ch = t + h * 256;
      const int r = ch >> 3, kc = ch & 7;
      unsigned long long ga = (unsigned long long)(Wb + (size_t)(n0 + r) * K + k0 + kc * 8);
      unsigned lds = (unsigned)(size_t)(lb + r * LDSROW + kc * 8);
      asm volatile("global_load_async_to_lds_b128 %0, %1, off" :: "v"(lds), "v"(ga) : "memory");
    }
  };

  v8f acc[2][2];
  #pragma unroll
  for (int mi = 0; mi < 2; ++mi)
    #pragma unroll
    for (int ni = 0; ni < 2; ++ni)
      acc[mi][ni] = (v8f){0.f, 0.f, 0.f, 0.f, 0.f, 0.f, 0.f, 0.f};

  auto compute = [&](int buf) {
    const bf16* la = sA + buf * ATILE_E;
    const bf16* lb = sB + buf * BTILE_E;
    #pragma unroll
    for (int ks = 0; ks < 64; ks += 32) {
      Frag fa[2], fb[2];
      // A fragments (16x32): lanes 0-15 K 0-7/16-23, lanes 16-31 K 8-15/24-31
      #pragma unroll
      for (int mi = 0; mi < 2; ++mi) {
        const int am = wm * 32 + mi * 16 + lr;
        #pragma unroll
        for (int p = 0; p < 8; ++p) {
          const int klo = ks + (((p & 4) << 2) | ((p & 3) << 1) | (lh << 3));
          fa[mi].u[p] = *reinterpret_cast<const unsigned int*>(&la[am * LDSROW + klo]);
        }
      }
      // B fragments (32x16): lane = N column, half-wave = K group
      #pragma unroll
      for (int ni = 0; ni < 2; ++ni) {
        const int bn = wn * 32 + ni * 16 + lr;
        #pragma unroll
        for (int p = 0; p < 8; ++p)
          fb[ni].u[p] = *reinterpret_cast<const unsigned int*>(&lb[bn * LDSROW + ks + (((p & 4) << 2) | ((p & 3) << 1) | (lh << 3))]);
      }
      #pragma unroll
      for (int mi = 0; mi < 2; ++mi)
        #pragma unroll
        for (int ni = 0; ni < 2; ++ni)
          acc[mi][ni] = __builtin_amdgcn_wmma_f32_16x16x32_bf16(
              false, fa[mi].v, false, fb[ni].v, (short)0, acc[mi][ni], false, false);
    }
  };

  const int steps = K >> 6;            // K / 64, always >= 2
  stage(0, 0);                         // prime the pipeline

  for (int i = 0; i < steps - 1; ++i) {
    stage((i + 1) << 6, (i + 1) & 1);  // next tile in flight
    asm volatile("s_wait_asynccnt 0x6" ::: "memory");   // current tile done
    __syncthreads();                   // all waves' tiles visible
    compute(i & 1);
    __syncthreads();                   // tile consumed; buffer reusable
  }
  asm volatile("s_wait_asynccnt 0x0" ::: "memory");
  __syncthreads();
  compute((steps - 1) & 1);

  // C/D layout: lane -> N = lane&15; VGPR r -> M = r + 8*(lane>=16)
  #pragma unroll
  for (int mi = 0; mi < 2; ++mi) {
    const int gm = m0 + wm * 32 + mi * 16 + lh * 8;
    #pragma unroll
    for (int ni = 0; ni < 2; ++ni) {
      const int gn = n0 + wn * 32 + ni * 16 + lr;
      const float bv = Bb[gn];
      #pragma unroll
      for (int r = 0; r < 8; ++r) {
        Ob[(size_t)(gm + r) * oBatch + gn] = f2bf(acc[mi][ni][r] + bv);
      }
    }
  }
}

// ---------------------------------------------------------------------------
// Final: channel-reduce (8->1), tanh, mask-fuse.  d_out = [fused | ori] f32.
// ---------------------------------------------------------------------------
__global__ void k_final(const bf16* __restrict__ u,
                        const float* __restrict__ ow, const float* __restrict__ ob,
                        const float* __restrict__ s2, float* __restrict__ out, int total) {
  int idx = blockIdx.x * blockDim.x + threadIdx.x;
  if (idx >= total) return;
  int bb = idx >> 10, l = idx & 1023;
  const bf16* ub = u + (size_t)bb * 8 * 1024 + l;
  float y = ob[0];
  #pragma unroll
  for (int c = 0; c < 8; ++c) y += bf2f(ub[(size_t)c * 1024]) * ow[c];
  float ori  = tanhf(y);
  const float* s2b = s2 + (size_t)bb * 8 * 1024;
  float ndvi = s2b[l];                 // channel 0
  float mask = s2b[7 * 1024 + l];      // channel -1
  out[idx]                 = ndvi * mask + ori * (1.f - mask);
  out[(size_t)total + idx] = ori;
}

// ---------------------------------------------------------------------------
// Host orchestration
// ---------------------------------------------------------------------------
extern "C" void kernel_launch(void* const* d_in, const int* in_sizes, int n_in,
                              void* d_out, int out_size, void* d_ws, size_t ws_size,
                              hipStream_t stream) {
  (void)in_sizes; (void)n_in; (void)out_size; (void)ws_size;
  auto F = [&](int i) -> const float* { return (const float*)d_in[i]; };

  // ---- workspace carve (bf16 buffers, 256B aligned) ----
  char* ws = (char*)d_ws;
  size_t off = 0;
  auto carve = [&](size_t elems) -> bf16* {
    bf16* p = (bf16*)(ws + off);
    off += ((elems * sizeof(bf16) + 255) / 256) * 256;
    return p;
  };
  const int Ls[4] = {1024, 512, 256, 128};
  bf16* x1    = carve((size_t)Bn * 4 * 1024);
  bf16* x2    = carve((size_t)Bn * 4 * 512);
  bf16* x3    = carve((size_t)Bn * 4 * 256);
  bf16* x4    = carve((size_t)Bn * 4 * 128);
  bf16* h1    = carve((size_t)Bn * 2 * 1536);
  bf16* h2    = carve((size_t)Bn * 8 * 1536);
  bf16* cat34 = carve((size_t)Bn * 8 * 384);   // [e3 | e4] along length
  bf16* cat2  = carve((size_t)Bn * 8 * 768);   // [e2 | d4]
  bf16* cat1  = carve((size_t)Bn * 8 * 1536);  // [e1 | d3]
  bf16* uout  = carve((size_t)Bn * 8 * 1024);
  bf16* wd[8];
  for (int s = 0; s < 2; ++s)
    for (int j = 0; j < 4; ++j)
      wd[s * 4 + j] = carve((size_t)4 * Ls[j] * Ls[j]);
  bf16* wu3 = carve((size_t)8 * 256 * 384);
  bf16* wu2 = carve((size_t)8 * 512 * 768);
  bf16* wu1 = carve((size_t)8 * 1024 * 1536);

  // ---- convert lin_w weights to bf16 (one pass, lives in L2 afterwards) ----
  auto cvt = [&](const float* src, bf16* dst, size_t n) {
    k_f2bf<<<(unsigned)((n + 255) / 256), 256, 0, stream>>>(src, dst, (int)n);
  };
  for (int s = 0; s < 2; ++s)
    for (int j = 0; j < 4; ++j)
      cvt(F(6 + (s * 4 + j) * 6 + 4), wd[s * 4 + j], (size_t)4 * Ls[j] * Ls[j]);
  cvt(F(54 + 4), wu3, (size_t)8 * 256 * 384);
  cvt(F(60 + 4), wu2, (size_t)8 * 512 * 768);
  cvt(F(66 + 4), wu1, (size_t)8 * 1024 * 1536);

  // ---- encoder (two streams, x buffers reused) ----
  bf16* xs[4] = {x1, x2, x3, x4};
  struct Dst { bf16* base; int Lbuf; int lo; };
  const Dst dst[4] = {{cat1, 1536, 0}, {cat2, 768, 0}, {cat34, 384, 0}, {cat34, 384, 256}};

  for (int s = 0; s < 2; ++s) {
    const int Cin = s ? 8 : 6;
    int tot = Bn * 1024;
    k_conv1x1_relu<<<(tot + 255) / 256, 256, 0, stream>>>(
        F(s), F(2 + 2 * s), F(3 + 2 * s), x1, Cin, 1024, tot);
    for (int j = 0; j < 4; ++j) {
      const int L = Ls[j];
      const int pb = 6 + (s * 4 + j) * 6;
      tot = Bn * L;
      k_conv5_ac1<<<(tot + 255) / 256, 256, 0, stream>>>(xs[j], F(pb + 0), F(pb + 1), h1, 4, L, tot);
      k_conv5_ac2<<<(tot + 255) / 256, 256, 0, stream>>>(h1, F(pb + 2), F(pb + 3), h2, 4, L, tot);
      dim3 g(Bn / 128, L / 64, 4);
      bf16* outp = dst[j].base + (size_t)(s * 4) * dst[j].Lbuf + dst[j].lo;
      k_bmm_wmma<<<g, 256, 0, stream>>>(h2, 4 * L, L,
                                        wd[s * 4 + j], L * L,
                                        F(pb + 5), L,
                                        outp, 8 * dst[j].Lbuf, dst[j].Lbuf,
                                        L, L);
      if (j < 3) {
        int tp = Bn * 4 * (L / 2);
        k_maxpool<<<(tp + 255) / 256, 256, 0, stream>>>(xs[j], xs[j + 1], 4, L, tp);
      }
    }
  }

  // ---- decoder ----
  struct Up { const bf16* inx; int Lin; int Nout; bf16* w; int pb; bf16* outBase; int Lbuf; int lo; };
  const Up ups[3] = {
      {cat34, 384, 256, wu3, 54, cat2, 768, 512},
      {cat2, 768, 512, wu2, 60, cat1, 1536, 1024},
      {cat1, 1536, 1024, wu1, 66, uout, 1024, 0},
  };
  for (int u = 0; u < 3; ++u) {
    const Up& U = ups[u];
    int tot = Bn * U.Lin;
    k_conv5_ac1<<<(tot + 255) / 256, 256, 0, stream>>>(U.inx, F(U.pb + 0), F(U.pb + 1), h1, 8, U.Lin, tot);
    k_conv5_ac2<<<(tot + 255) / 256, 256, 0, stream>>>(h1, F(U.pb + 2), F(U.pb + 3), h2, 8, U.Lin, tot);
    dim3 g(Bn / 128, U.Nout / 64, 8);
    bf16* outp = U.outBase + U.lo;
    k_bmm_wmma<<<g, 256, 0, stream>>>(h2, 8 * U.Lin, U.Lin,
                                      U.w, U.Nout * U.Lin,
                                      F(U.pb + 5), U.Nout,
                                      outp, 8 * U.Lbuf, U.Lbuf,
                                      U.Lin, U.Nout);
  }

  // ---- output head ----
  int tot = Bn * 1024;
  k_final<<<(tot + 255) / 256, 256, 0, stream>>>(uout, F(72), F(73), F(1), (float*)d_out, tot);
}
